// LowMemoryAttention_41386304864565
// MI455X (gfx1250) — matrix operand order repaired and checked by reading, not compiled, not run on hardware
//
#include <hip/hip_runtime.h>
#include <hip/hip_bf16.h>

typedef __bf16 v16bf __attribute__((ext_vector_type(16)));
typedef __bf16 v8bf  __attribute__((ext_vector_type(8)));
typedef __bf16 v4bf  __attribute__((ext_vector_type(4)));
typedef float  v8f   __attribute__((ext_vector_type(8)));
typedef float  v4f   __attribute__((ext_vector_type(4)));

#define NDIM 4096
#define CDIM 256
#define HEADS 8
#define DDIM 64
#define HDDIM 512

#if defined(__gfx1250__) && \
    __has_builtin(__builtin_amdgcn_global_load_async_to_lds_b128)
#define HAVE_ASYNC_LDS 1
typedef int v4i_vs __attribute__((__vector_size__(4 * sizeof(int))));
typedef __attribute__((address_space(1))) v4i_vs gas_int4;
typedef __attribute__((address_space(3))) v4i_vs las_int4;
#else
#define HAVE_ASYNC_LDS 0
#endif

__device__ inline void wait_async0() {
#if __has_builtin(__builtin_amdgcn_s_wait_asynccnt)
  __builtin_amdgcn_s_wait_asynccnt(0);
#elif defined(__gfx1250__)
  asm volatile("s_wait_asynccnt 0x0" ::: "memory");
#endif
}

__device__ inline v8f zero_v8f() { v8f z = {}; return z; }

__device__ inline v8f wmma_bf16(v16bf a, v16bf b, v8f c) {
  // 8 args: (neg_a, A, neg_b, B, c_mod, C, reuse_a, reuse_b)
  return __builtin_amdgcn_wmma_f32_16x16x32_bf16(false, a, false, b, (short)0, c,
                                                 false, false);
}

// A fragment, 16x32 bf16: lane L -> row L%16; K chunks at half*8 and 16+half*8.
__device__ inline v16bf ldfragA(const __bf16* base, int ldk, int l15, int hf) {
  const __bf16* p = base + l15 * ldk + hf * 8;
  union { v16bf v; v8bf q[2]; } u;
  u.q[0] = *(const v8bf*)p;
  u.q[1] = *(const v8bf*)(p + 16);
  return u.v;
}

// B fragment, 32x16 bf16 stored [col][K]: lane L -> col L%16; K chunks at half*8 and 16+half*8.
__device__ inline v16bf ldfragB(const __bf16* base, int ldk, int l15, int hf) {
  const __bf16* p = base + l15 * ldk + hf * 8;
  union { v16bf v; v8bf q[2]; } u;
  u.q[0] = *(const v8bf*)p;
  u.q[1] = *(const v8bf*)(p + 16);
  return u.v;
}

__device__ inline void load4_to_bf16(const float* p, __bf16* d) {
  v4f a = *(const v4f*)p;
  v4bf c;
  c[0] = (__bf16)a[0]; c[1] = (__bf16)a[1];
  c[2] = (__bf16)a[2]; c[3] = (__bf16)a[3];
  *(v4bf*)d = c;
}
__device__ inline void load4_to_bf16(const __bf16* p, __bf16* d) {
  *(v4bf*)d = *(const v4bf*)p;
}

// ---------------------------------------------------------------------------
// Tiled WMMA GEMM:  Y[M][O] = X[M][K] * W[O][K]^T   (+ epilogue)
// Block tile 128x128, 256 threads (8 waves, 4m x 2n), K-chunks of 64 in LDS.
// mode 0: outB = bf16(val*scale)          (q / k projections)
// mode 2: outB[col*ldo+row] = bf16(val)   (v stored transposed [HD][N])
// mode 3: outF = sigmoid(val + bvec[col]) (gating, f32)
// mode 4: outF = val + bvec[col]          (final output, f32)
// ---------------------------------------------------------------------------
template<typename AT>
__global__ __launch_bounds__(256)
void wmma_gemm(const AT* __restrict__ A, const float* __restrict__ W,
               int M, int O, int K, int mode, float scale,
               const float* __restrict__ bvec,
               __bf16* __restrict__ outB, float* __restrict__ outF, int ldo) {
  constexpr int KC  = 64;
  constexpr int LDK = 88;  // 64 + 24 pad (bank-conflict break, keeps 16B align)
  __shared__ __align__(16) __bf16 sA[128 * LDK];
  __shared__ __align__(16) __bf16 sB[128 * LDK];

  const int tid  = threadIdx.x;
  const int lane = tid & 31;
  const int wv   = tid >> 5;
  const int l15  = lane & 15;
  const int hf   = lane >> 4;
  const int wm   = wv & 3;   // 4 m-blocks of 32 rows
  const int wn   = wv >> 2;  // 2 n-blocks of 64 cols
  const int m0   = blockIdx.x * 128;
  const int n0   = blockIdx.y * 128;

  v8f acc[2][4];
#pragma unroll
  for (int i = 0; i < 2; ++i)
#pragma unroll
    for (int j = 0; j < 4; ++j) acc[i][j] = zero_v8f();

  const int lrow = tid >> 4;        // 0..15
  const int lcol = (tid & 15) * 4;  // 0..60

  for (int kc = 0; kc < K; kc += KC) {
#pragma unroll
    for (int rp = 0; rp < 128; rp += 16) {
      int r = rp + lrow;
      load4_to_bf16(A + (size_t)(m0 + r) * K + kc + lcol, &sA[r * LDK + lcol]);
      load4_to_bf16(W + (size_t)(n0 + r) * K + kc + lcol, &sB[r * LDK + lcol]);
    }
    __syncthreads();
#pragma unroll
    for (int ks = 0; ks < KC; ks += 32) {
      v16bf af[2], bfr[4];
#pragma unroll
      for (int mf = 0; mf < 2; ++mf)
        af[mf] = ldfragA(&sA[(wm * 32 + mf * 16) * LDK + ks], LDK, l15, hf);
#pragma unroll
      for (int nf = 0; nf < 4; ++nf)
        bfr[nf] = ldfragB(&sB[(wn * 64 + nf * 16) * LDK + ks], LDK, l15, hf);
#pragma unroll
      for (int mf = 0; mf < 2; ++mf)
#pragma unroll
        for (int nf = 0; nf < 4; ++nf)
          acc[mf][nf] = wmma_bf16(af[mf], bfr[nf], acc[mf][nf]);
    }
    __syncthreads();
  }

  // Epilogue: C/D layout -> row = half*8+e (lanes 0-15 rows 0-7), col = lane%16
#pragma unroll
  for (int mf = 0; mf < 2; ++mf)
#pragma unroll
    for (int nf = 0; nf < 4; ++nf)
#pragma unroll
      for (int e = 0; e < 8; ++e) {
        int row = m0 + wm * 32 + mf * 16 + hf * 8 + e;
        int col = n0 + wn * 64 + nf * 16 + l15;
        float val = acc[mf][nf][e];
        if (mode == 0) {
          outB[(size_t)row * ldo + col] = (__bf16)(val * scale);
        } else if (mode == 2) {
          outB[(size_t)col * ldo + row] = (__bf16)val;
        } else if (mode == 3) {
          float x = val + bvec[col];
          outF[(size_t)row * ldo + col] = 1.0f / (1.0f + __expf(-x));
        } else {
          outF[(size_t)row * ldo + col] = val + bvec[col];
        }
      }
}

// ---------------------------------------------------------------------------
// Flash attention with gating.
// Grid: N/16 blocks, 256 threads = 8 waves; wave w handles head w for 16 rows.
// Bias tile (16 x 64 f32) is async-copied (GLOBAL_LOAD_ASYNC_TO_LDS_B128,
// ASYNCcnt) into double-buffered LDS one kv-step ahead and shared by all 8
// heads (keeps bias HBM traffic at the optimal 64 MB, latency hidden by WMMA).
// ---------------------------------------------------------------------------
__global__ __launch_bounds__(256)
void attn_kernel(const __bf16* __restrict__ qb, const __bf16* __restrict__ kb,
                 const __bf16* __restrict__ vt, const float* __restrict__ bias,
                 const float* __restrict__ gf, __bf16* __restrict__ og) {
  constexpr int LB = 68;  // bias LDS stride (f32): 68*4 = 272B = 17*16B rows
  constexpr int LP = 88;  // P scratch stride (bf16), 16B-aligned rows
  __shared__ __align__(16) float  sBias[2][16 * LB];
  __shared__ __align__(16) __bf16 sP[HEADS][16 * LP];

  const int tid  = threadIdx.x;
  const int lane = tid & 31;
  const int h    = tid >> 5;  // head
  const int l15  = lane & 15;
  const int hf   = lane >> 4;
  const int q0   = blockIdx.x * 16;

  // Persistent Q fragments (16 rows x D=64 -> two 16x32 A-frags)
  v16bf aq[2];
  const __bf16* qrow = qb + (size_t)(q0 + l15) * HDDIM + h * DDIM;
#pragma unroll
  for (int kc = 0; kc < 2; ++kc) {
    union { v16bf v; v8bf q[2]; } u;
    const __bf16* p = qrow + kc * 32 + hf * 8;
    u.q[0] = *(const v8bf*)p;
    u.q[1] = *(const v8bf*)(p + 16);
    aq[kc] = u.v;
  }

  v8f o[4];
#pragma unroll
  for (int dc = 0; dc < 4; ++dc) o[dc] = zero_v8f();
  float mst[8], lst[8];
#pragma unroll
  for (int e = 0; e < 8; ++e) { mst[e] = -__builtin_inff(); lst[e] = 0.0f; }

  const int brow = tid >> 4;        // 0..15
  const int bcol = (tid & 15) * 4;  // 0..60
  const float* bsrc = bias + (size_t)(q0 + brow) * NDIM + bcol;

  constexpr int NT = NDIM / 64;

#if HAVE_ASYNC_LDS
  // Prologue: async-copy bias tile 0 into buffer 0 (memory -> LDS, no VGPRs).
  __builtin_amdgcn_global_load_async_to_lds_b128(
      (gas_int4*)(void*)(bsrc), (las_int4*)(void*)&sBias[0][brow * LB + bcol],
      0, 0);
#endif

  for (int kt = 0; kt < NT; ++kt) {
    const int kv0 = kt * 64;
    float* bb = sBias[kt & 1];
#if HAVE_ASYNC_LDS
    wait_async0();     // own chunks of tile kt are in LDS
    __syncthreads();   // all waves' chunks done; all done reading tile kt-1
    if (kt + 1 < NT) {
      // Prefetch next bias tile into the other buffer while we do WMMA work.
      __builtin_amdgcn_global_load_async_to_lds_b128(
          (gas_int4*)(void*)(bsrc + (kt + 1) * 64),
          (las_int4*)(void*)&sBias[(kt + 1) & 1][brow * LB + bcol], 0, 0);
    }
#else
    {
      v4f bv = *(const v4f*)(bsrc + kv0);
      bb[brow * LB + bcol + 0] = bv[0];
      bb[brow * LB + bcol + 1] = bv[1];
      bb[brow * LB + bcol + 2] = bv[2];
      bb[brow * LB + bcol + 3] = bv[3];
    }
    __syncthreads();
#endif

    // S = Q K^T  (16 x 64)
    v8f s[4];
#pragma unroll
    for (int nb = 0; nb < 4; ++nb) s[nb] = zero_v8f();
#pragma unroll
    for (int kc = 0; kc < 2; ++kc) {
#pragma unroll
      for (int nb = 0; nb < 4; ++nb) {
        const __bf16* p = kb + (size_t)(kv0 + nb * 16 + l15) * HDDIM +
                          h * DDIM + kc * 32 + hf * 8;
        union { v16bf v; v8bf q[2]; } u;
        u.q[0] = *(const v8bf*)p;
        u.q[1] = *(const v8bf*)(p + 16);
        s[nb] = wmma_bf16(aq[kc], u.v, s[nb]);
      }
    }

    // bias add + online softmax (row = hf*8+e, 16 lanes/half share a row)
#pragma unroll
    for (int e = 0; e < 8; ++e) {
      float mb = -__builtin_inff();
#pragma unroll
      for (int nb = 0; nb < 4; ++nb) {
        float x = s[nb][e] + bb[(hf * 8 + e) * LB + nb * 16 + l15];
        s[nb][e] = x;
        mb = fmaxf(mb, x);
      }
      mb = fmaxf(mb, __shfl_xor(mb, 1));
      mb = fmaxf(mb, __shfl_xor(mb, 2));
      mb = fmaxf(mb, __shfl_xor(mb, 4));
      mb = fmaxf(mb, __shfl_xor(mb, 8));
      float mnew = fmaxf(mst[e], mb);
      float sc = __expf(mst[e] - mnew);
      float r = 0.0f;
#pragma unroll
      for (int nb = 0; nb < 4; ++nb) {
        float p = __expf(s[nb][e] - mnew);
        r += p;
        sP[h][(hf * 8 + e) * LP + nb * 16 + l15] = (__bf16)p;
      }
      r += __shfl_xor(r, 1);
      r += __shfl_xor(r, 2);
      r += __shfl_xor(r, 4);
      r += __shfl_xor(r, 8);
      lst[e] = lst[e] * sc + r;
      mst[e] = mnew;
#pragma unroll
      for (int dc = 0; dc < 4; ++dc) o[dc][e] *= sc;
    }
#if __has_builtin(__builtin_amdgcn_s_wait_dscnt)
    __builtin_amdgcn_s_wait_dscnt(0);
#elif defined(__gfx1250__)
    asm volatile("s_wait_dscnt 0x0" ::: "memory");
#endif

    // O += P V   (A-frags of P from LDS scratch, B-frags of V^T from global)
#pragma unroll
    for (int kc2 = 0; kc2 < 2; ++kc2) {
      union { v16bf v; v8bf q[2]; } ua;
      const __bf16* pp = &sP[h][l15 * LP + kc2 * 32 + hf * 8];
      ua.q[0] = *(const v8bf*)pp;
      ua.q[1] = *(const v8bf*)(pp + 16);
#pragma unroll
      for (int dc = 0; dc < 4; ++dc) {
        const __bf16* pv = vt + (size_t)(h * DDIM + dc * 16 + l15) * NDIM +
                           kv0 + kc2 * 32 + hf * 8;
        union { v16bf v; v8bf q[2]; } ub;
        ub.q[0] = *(const v8bf*)pv;
        ub.q[1] = *(const v8bf*)(pv + 16);
        o[dc] = wmma_bf16(ua.v, ub.v, o[dc]);
      }
    }
  }

  // Normalize, gate, store bf16
#pragma unroll
  for (int dc = 0; dc < 4; ++dc)
#pragma unroll
    for (int e = 0; e < 8; ++e) {
      int row = q0 + hf * 8 + e;
      int col = h * DDIM + dc * 16 + l15;
      float val = o[dc][e] / lst[e];
      float gv = gf[(size_t)row * HDDIM + col];
      og[(size_t)row * HDDIM + col] = (__bf16)(val * gv);
    }
}

extern "C" void kernel_launch(void* const* d_in, const int* in_sizes, int n_in,
                              void* d_out, int out_size, void* d_ws,
                              size_t ws_size, hipStream_t stream) {
  const float* q_x  = (const float*)d_in[0];
  const float* k_x  = (const float*)d_in[1];
  const float* v_x  = (const float*)d_in[2];
  const float* bias = (const float*)d_in[3];
  const float* wq   = (const float*)d_in[4];
  const float* wk   = (const float*)d_in[5];
  const float* wv   = (const float*)d_in[6];
  const float* wg   = (const float*)d_in[7];
  const float* bg   = (const float*)d_in[8];
  const float* wo   = (const float*)d_in[9];
  const float* bo   = (const float*)d_in[10];

  char* ws = (char*)d_ws;
  __bf16* qbf = (__bf16*)ws;                       // [N][HD] bf16, 4 MB
  __bf16* kbf = qbf + (size_t)NDIM * HDDIM;        // [N][HD] bf16, 4 MB
  __bf16* vtb = kbf + (size_t)NDIM * HDDIM;        // [HD][N] bf16, 4 MB
  __bf16* ogb = vtb + (size_t)NDIM * HDDIM;        // [N][HD] bf16, 4 MB
  float*  gfp = (float*)(ogb + (size_t)NDIM * HDDIM);  // [N][HD] f32, 8 MB

  dim3 blk(256);
  dim3 g1(NDIM / 128, HDDIM / 128);
  // q = (x wq^T)/8 -> bf16
  wmma_gemm<float><<<g1, blk, 0, stream>>>(q_x, wq, NDIM, HDDIM, CDIM, 0,
                                           0.125f, nullptr, qbf, nullptr, HDDIM);
  // k -> bf16
  wmma_gemm<float><<<g1, blk, 0, stream>>>(k_x, wk, NDIM, HDDIM, CDIM, 0, 1.0f,
                                           nullptr, kbf, nullptr, HDDIM);
  // v -> bf16, transposed [HD][N]
  wmma_gemm<float><<<g1, blk, 0, stream>>>(v_x, wv, NDIM, HDDIM, CDIM, 2, 1.0f,
                                           nullptr, vtb, nullptr, NDIM);
  // g = sigmoid(x wg^T + bg) -> f32
  wmma_gemm<float><<<g1, blk, 0, stream>>>(q_x, wg, NDIM, HDDIM, CDIM, 3, 1.0f,
                                           bg, nullptr, gfp, HDDIM);
  // flash attention + gating -> og bf16
  attn_kernel<<<dim3(NDIM / 16), blk, 0, stream>>>(qbf, kbf, vtb, bias, gfp,
                                                   ogb);
  // out = og wo^T + bo -> f32
  dim3 g2(NDIM / 128, CDIM / 128);
  wmma_gemm<__bf16><<<g2, blk, 0, stream>>>(ogb, wo, NDIM, CDIM, HDDIM, 4, 1.0f,
                                            bo, nullptr, (float*)d_out, CDIM);
}
